// IRENE_72739566125852
// MI455X (gfx1250) — matrix operand order repaired and checked by reading, not compiled, not run on hardware
//
#include <hip/hip_runtime.h>
#include <hip/hip_bf16.h>

typedef __attribute__((ext_vector_type(8)))  __bf16 v8bf;
typedef __attribute__((ext_vector_type(16))) __bf16 v16bf;
typedef __attribute__((ext_vector_type(8)))  float  v8f;

#define BB   32
#define NN   256
#define KKN  8
#define DD   1024
#define OUTD 1024
#define HIDD 2048
#define ROWS (BB*NN)      /* 8192 */
#define EPSB 1e-5f

#define TM 128            /* block tile M */
#define TN 128            /* block tile N */
#define TK 32             /* K step */
#define LDA 40            /* padded LDS row stride (elements) */
#define LDB 40
#define RBLK 32           /* row blocks for column stats */

// ---------------------------------------------------------------------------
// Kernel 1: entity = mean_k(neigh*mask); X = [self || entity] in bf16
// ---------------------------------------------------------------------------
__global__ __launch_bounds__(256)
void irene_build_x(const float* __restrict__ selfv,
                   const float* __restrict__ neigh,
                   const float* __restrict__ masks,
                   __bf16* __restrict__ X) {
    const int r = blockIdx.x;
    const int t = threadIdx.x;

    float m[KKN];
#pragma unroll
    for (int k = 0; k < KKN; ++k) m[k] = masks[r * KKN + k] * (1.0f / KKN);

    {
        float4 v = ((const float4*)(selfv + (size_t)r * DD))[t];
        __bf16* xo = X + (size_t)r * 3 * DD + t * 4;
        xo[0] = (__bf16)v.x; xo[1] = (__bf16)v.y;
        xo[2] = (__bf16)v.z; xo[3] = (__bf16)v.w;
    }
    const size_t nb = (size_t)r * KKN * 2 * DD;
#pragma unroll
    for (int seg = 0; seg < 2; ++seg) {
        const int j = (t + seg * 256) * 4;
        float ax = 0.f, ay = 0.f, az = 0.f, aw = 0.f;
#pragma unroll
        for (int k = 0; k < KKN; ++k) {
            float4 v = *(const float4*)(neigh + nb + (size_t)k * 2 * DD + j);
            ax += v.x * m[k]; ay += v.y * m[k];
            az += v.z * m[k]; aw += v.w * m[k];
        }
        __bf16* xo = X + (size_t)r * 3 * DD + DD + j;
        xo[0] = (__bf16)ax; xo[1] = (__bf16)ay;
        xo[2] = (__bf16)az; xo[3] = (__bf16)aw;
    }
}

// ---------------------------------------------------------------------------
// fp32 -> bf16 conversion WITH transpose: in [Kt x N] -> out [N x Kt]
// tiled via LDS so both global sides stay coalesced. one-time cost.
// ---------------------------------------------------------------------------
__global__ __launch_bounds__(256)
void irene_cvt_transpose(const float* __restrict__ in, __bf16* __restrict__ out,
                         int Kt, int N) {
    __shared__ float tile[32][33];
    const int kb = blockIdx.x * 32;
    const int nb = blockIdx.y * 32;
    const int tx = threadIdx.x & 31;
    const int ty = threadIdx.x >> 5;      // 8 rows per pass
#pragma unroll
    for (int i = 0; i < 32; i += 8)
        tile[ty + i][tx] = in[(size_t)(kb + ty + i) * N + nb + tx];
    __syncthreads();
#pragma unroll
    for (int i = 0; i < 32; i += 8)
        out[(size_t)(nb + ty + i) * Kt + kb + tx] = (__bf16)tile[tx][ty + i];
}

// ---------------------------------------------------------------------------
// Tiled bf16 WMMA GEMM:  out = A[M x Kt] * Wt^T + bias,  Wt is [N x Kt]
// Block tile 128x128, 8 waves as 4(M) x 2(N); each wave: 2 M-tiles x 4 N-tiles
// => 8 WMMAs per K-step from 6 fragment loads. Weights pre-transposed, so A
// and B staging are symmetric: 4 contiguous ds_store_b128 per thread.
// A-fragment per ISA: lanes<16 hold K 0-7,16-23 of row M=lane&15; lanes>=16
// hold K 8-15,24-31. B fragment: lane (col=lane&15) reads the same K order
// (lanes<16: K 0-7,16-23, lanes>=16: K 8-15,24-31) from the [n][k] LDS tile.
// ---------------------------------------------------------------------------
template<bool WF32, bool WBF16>
__global__ __launch_bounds__(256)
void irene_gemm_bf16(const __bf16* __restrict__ A, const __bf16* __restrict__ Wt,
                     const float* __restrict__ bias,
                     float* __restrict__ outF, __bf16* __restrict__ outB,
                     int Kt, int N) {
    __shared__ __bf16 Alds[TM * LDA];
    __shared__ __bf16 Blds[TN * LDB];

    const int t     = threadIdx.x;
    const int wave  = t >> 5;
    const int lane  = t & 31;
    const int half  = lane >> 4;     // 0 or 1
    const int lrow  = lane & 15;
    const int waveM = wave & 3;      // 0..3
    const int waveN = wave >> 2;     // 0..1
    const int m0 = blockIdx.x * TM;
    const int n0 = blockIdx.y * TN;

    v8f acc[2][4];
#pragma unroll
    for (int mt = 0; mt < 2; ++mt)
#pragma unroll
        for (int nt = 0; nt < 4; ++nt)
#pragma unroll
            for (int e = 0; e < 8; ++e) acc[mt][nt][e] = 0.f;

    // prefetch registers: 2 A chunks + 2 B chunks per thread per K-tile
    v8bf aR0, aR1, bR0, bR1;
    const int row0 = t >> 2;               // 0..63
    const int part = (t & 3) * 8;

    auto gload = [&](int k0) {
        aR0 = *(const v8bf*)(A  + (size_t)(m0 + row0) * Kt + k0 + part);
        aR1 = *(const v8bf*)(A  + (size_t)(m0 + 64 + row0) * Kt + k0 + part);
        bR0 = *(const v8bf*)(Wt + (size_t)(n0 + row0) * Kt + k0 + part);
        bR1 = *(const v8bf*)(Wt + (size_t)(n0 + 64 + row0) * Kt + k0 + part);
    };
    auto lstore = [&]() {
        *(v8bf*)(&Alds[row0 * LDA + part]) = aR0;
        *(v8bf*)(&Alds[(64 + row0) * LDA + part]) = aR1;
        *(v8bf*)(&Blds[row0 * LDB + part]) = bR0;
        *(v8bf*)(&Blds[(64 + row0) * LDB + part]) = bR1;
    };

    gload(0);
    lstore();
    __syncthreads();

    const int kbA = half * 8;
    const int kbB = half * 8;

    for (int k0 = 0; k0 < Kt; k0 += TK) {
        const bool has_next = (k0 + TK) < Kt;
        if (has_next) gload(k0 + TK);   // issue global loads early

        // load all fragments, then fire 8 WMMAs back-to-back
        v16bf afr[2], bfr[4];
#pragma unroll
        for (int mt = 0; mt < 2; ++mt) {
            const int row = waveM * 32 + mt * 16 + lrow;
            v8bf lo = *(const v8bf*)(&Alds[row * LDA + kbA]);
            v8bf hi = *(const v8bf*)(&Alds[row * LDA + kbA + 16]);
            afr[mt] = __builtin_shufflevector(lo, hi,
                        0,1,2,3,4,5,6,7,8,9,10,11,12,13,14,15);
        }
#pragma unroll
        for (int nt = 0; nt < 4; ++nt) {
            const int coln = waveN * 64 + nt * 16 + lrow;
            v8bf lo = *(const v8bf*)(&Blds[coln * LDB + kbB]);
            v8bf hi = *(const v8bf*)(&Blds[coln * LDB + kbB + 16]);
            bfr[nt] = __builtin_shufflevector(lo, hi,
                        0,1,2,3,4,5,6,7,8,9,10,11,12,13,14,15);
        }
#pragma unroll
        for (int mt = 0; mt < 2; ++mt)
#pragma unroll
            for (int nt = 0; nt < 4; ++nt)
                acc[mt][nt] = __builtin_amdgcn_wmma_f32_16x16x32_bf16(
                    false, afr[mt], false, bfr[nt], (short)0, acc[mt][nt],
                    false, false);

        if (has_next) {
            __syncthreads();   // everyone done reading this tile
            lstore();          // waits on global loads here, after compute
            __syncthreads();
        }
    }

    // epilogue: bias add + store (compile-time selected)
#pragma unroll
    for (int mt = 0; mt < 2; ++mt) {
#pragma unroll
        for (int nt = 0; nt < 4; ++nt) {
            const int col = n0 + waveN * 64 + nt * 16 + lrow;
            const float bv = bias[col];
#pragma unroll
            for (int v = 0; v < 8; ++v) {
                const int row = m0 + waveM * 32 + mt * 16 + (half ? v + 8 : v);
                const float val = acc[mt][nt][v] + bv;
                if (WF32)  outF[(size_t)row * N + col] = val;
                if (WBF16) outB[(size_t)row * N + col] = (__bf16)val;
            }
        }
    }
}

// ---------------------------------------------------------------------------
// Column stats stage 1: partial sum / sumsq per (rowblock, col) — coalesced.
// ---------------------------------------------------------------------------
__global__ __launch_bounds__(256)
void irene_colpart(const float* __restrict__ X, int N,
                   float* __restrict__ P1, float* __restrict__ P2) {
    const int col = blockIdx.x * 256 + threadIdx.x;
    const int r0  = blockIdx.y * (ROWS / RBLK);
    float s = 0.f, q = 0.f;
    for (int r = r0; r < r0 + (ROWS / RBLK); ++r) {
        float v = X[(size_t)r * N + col];
        s += v; q += v * v;
    }
    P1[(size_t)blockIdx.y * N + col] = s;
    P2[(size_t)blockIdx.y * N + col] = q;
}

__global__ __launch_bounds__(256)
void irene_colfin(const float* __restrict__ P1, const float* __restrict__ P2, int N,
                  float* __restrict__ mu, float* __restrict__ var) {
    const int col = blockIdx.x * 256 + threadIdx.x;
    float s = 0.f, q = 0.f;
#pragma unroll
    for (int b = 0; b < RBLK; ++b) {
        s += P1[(size_t)b * N + col];
        q += P2[(size_t)b * N + col];
    }
    const float m = s * (1.0f / ROWS);
    mu[col]  = m;
    var[col] = q * (1.0f / ROWS) - m * m;
}

// ---------------------------------------------------------------------------
// BN + ReLU epilogues
// ---------------------------------------------------------------------------
__global__ __launch_bounds__(256)
void irene_bnrelu_bf16(const float* __restrict__ X, const float* __restrict__ mu,
                       const float* __restrict__ var, const float* __restrict__ g,
                       const float* __restrict__ be, __bf16* __restrict__ out, int N) {
    size_t i = (size_t)blockIdx.x * 256 + threadIdx.x;
    const size_t total = (size_t)ROWS * N;
    const size_t stride = (size_t)gridDim.x * 256;
    for (; i < total; i += stride) {
        const int c = (int)(i & (N - 1));
        float y = (X[i] - mu[c]) * rsqrtf(var[c] + EPSB) * g[c] + be[c];
        out[i] = (__bf16)(y > 0.f ? y : 0.f);
    }
}

__global__ __launch_bounds__(256)
void irene_bnrelu_f32(const float* __restrict__ X, const float* __restrict__ mu,
                      const float* __restrict__ var, const float* __restrict__ g,
                      const float* __restrict__ be, float* __restrict__ out, int N) {
    size_t i = (size_t)blockIdx.x * 256 + threadIdx.x;
    const size_t total = (size_t)ROWS * N;
    const size_t stride = (size_t)gridDim.x * 256;
    for (; i < total; i += stride) {
        const int c = (int)(i & (N - 1));
        float y = (X[i] - mu[c]) * rsqrtf(var[c] + EPSB) * g[c] + be[c];
        out[i] = y > 0.f ? y : 0.f;
    }
}

// ---------------------------------------------------------------------------
extern "C" void kernel_launch(void* const* d_in, const int* in_sizes, int n_in,
                              void* d_out, int out_size, void* d_ws, size_t ws_size,
                              hipStream_t stream) {
    const float* selfv = (const float*)d_in[0];
    const float* neigh = (const float*)d_in[1];
    const float* masks = (const float*)d_in[2];
    const float* W_agg = (const float*)d_in[3];
    const float* b_agg = (const float*)d_in[4];
    const float* W1    = (const float*)d_in[5];
    const float* b1    = (const float*)d_in[6];
    const float* g1    = (const float*)d_in[7];
    const float* be1   = (const float*)d_in[8];
    const float* W2    = (const float*)d_in[9];
    const float* b2    = (const float*)d_in[10];
    const float* g2    = (const float*)d_in[11];
    const float* be2   = (const float*)d_in[12];
    float* out = (float*)d_out;

    char* p = (char*)d_ws;
    __bf16* X      = (__bf16*)p; p += (size_t)ROWS * 3 * DD * 2;     // 48 MB
    __bf16* WaggT  = (__bf16*)p; p += (size_t)3 * DD * OUTD * 2;     //  6 MB [1024 x 3072]
    __bf16* W1T    = (__bf16*)p; p += (size_t)OUTD * HIDD * 2;       //  4 MB [2048 x 1024]
    __bf16* W2T    = (__bf16*)p; p += (size_t)HIDD * OUTD * 2;       //  4 MB [1024 x 2048]
    float*  F      = (float*) p; p += (size_t)ROWS * HIDD * 4;       // 64 MB
    __bf16* Act    = (__bf16*)p; p += (size_t)ROWS * HIDD * 2;       // 32 MB
    float*  P1     = (float*) p; p += (size_t)RBLK * HIDD * 4;
    float*  P2     = (float*) p; p += (size_t)RBLK * HIDD * 4;
    float*  mu1    = (float*) p; p += HIDD * 4;
    float*  var1   = (float*) p; p += HIDD * 4;
    float*  mu2    = (float*) p; p += OUTD * 4;
    float*  var2   = (float*) p; p += OUTD * 4;

    // 1) masked neighbor mean + concat, emit bf16 X [8192 x 3072]
    irene_build_x<<<ROWS, 256, 0, stream>>>(selfv, neigh, masks, X);

    // 2) convert + transpose weights to bf16 [N x Kt]
    irene_cvt_transpose<<<dim3(3 * DD / 32, OUTD / 32), 256, 0, stream>>>(
        W_agg, WaggT, 3 * DD, OUTD);
    irene_cvt_transpose<<<dim3(OUTD / 32, HIDD / 32), 256, 0, stream>>>(
        W1, W1T, OUTD, HIDD);
    irene_cvt_transpose<<<dim3(HIDD / 32, OUTD / 32), 256, 0, stream>>>(
        W2, W2T, HIDD, OUTD);

    // 3) GEMM1: h = X @ W_agg + b_agg -> bf16 Act [8192 x 1024]
    irene_gemm_bf16<false, true><<<dim3(ROWS / TM, OUTD / TN), 256, 0, stream>>>(
        X, WaggT, b_agg, nullptr, Act, 3 * DD, OUTD);

    // 4) GEMM2: t = h @ W1 + b1 -> fp32 F [8192 x 2048]
    irene_gemm_bf16<true, false><<<dim3(ROWS / TM, HIDD / TN), 256, 0, stream>>>(
        Act, W1T, b1, F, nullptr, OUTD, HIDD);

    // 5) column mean/var of t; h1 = relu(bn(t)) -> bf16 Act [8192 x 2048]
    irene_colpart<<<dim3(HIDD / 256, RBLK), 256, 0, stream>>>(F, HIDD, P1, P2);
    irene_colfin<<<HIDD / 256, 256, 0, stream>>>(P1, P2, HIDD, mu1, var1);
    irene_bnrelu_bf16<<<4096, 256, 0, stream>>>(F, mu1, var1, g1, be1, Act, HIDD);

    // 6) GEMM3: h2 = h1 @ W2 + b2 -> fp32 F [8192 x 1024]
    irene_gemm_bf16<true, false><<<dim3(ROWS / TM, OUTD / TN), 256, 0, stream>>>(
        Act, W2T, b2, F, nullptr, HIDD, OUTD);

    // 7) column mean/var of h2; out = relu(bn(h2)) -> fp32 d_out
    irene_colpart<<<dim3(OUTD / 256, RBLK), 256, 0, stream>>>(F, OUTD, P1, P2);
    irene_colfin<<<OUTD / 256, 256, 0, stream>>>(P1, P2, OUTD, mu2, var2);
    irene_bnrelu_f32<<<4096, 256, 0, stream>>>(F, mu2, var2, g2, be2, out, OUTD);
}
